// TensorProductConvLayer_14697378087508
// MI455X (gfx1250) — compile-verified, matrix operand order checked by reading, not run on hardware
//
#include <hip/hip_runtime.h>
#include <hip/hip_bf16.h>

// ---------------- problem constants ----------------
#define N_NODES   20000
#define N_EDGES   160000
#define MUL0      32
#define MUL1      16
#define EDGE_FEAT 128
#define HIDDEN    128
#define W_NUMEL   2560
#define FEAT      80              // MUL0 + 3*MUL1

// scalar coefficients (exact closed forms)
#define C0     0.14433756729740643f   // sqrt(1/48)
#define C0IS3  (1.0f/12.0f)           // sqrt(1/48)/sqrt(3)
#define C1IS3  0.125f                 // sqrt(3/64)/sqrt(3)
#define C1IS6  0.08838834764831845f   // sqrt(3/64)/sqrt(6) = 1/(8*sqrt2)

typedef _Float16 v16h __attribute__((ext_vector_type(16)));
typedef _Float16 v2h  __attribute__((ext_vector_type(2)));
typedef float    v8f  __attribute__((ext_vector_type(8)));
typedef unsigned int v4u __attribute__((ext_vector_type(4)));
typedef int      v8i  __attribute__((ext_vector_type(8)));
typedef int      v4i  __attribute__((ext_vector_type(4)));

// workspace layout (bytes)
#define ACC_BYTES   (size_t)(N_NODES * FEAT * 4)        // 6,400,000
#define CNT_OFF     ACC_BYTES
#define CNT_BYTES   (size_t)(N_NODES * 4)               //    80,000
#define W1F_OFF     (CNT_OFF + CNT_BYTES)               // 6,480,000
#define W1F_HALFS   (8 * 4 * 32 * 16)                   //    16,384 halfs
#define W2F_OFF     (W1F_OFF + (size_t)W1F_HALFS * 2)   // 6,512,768
#define W2F_HALFS   (160 * 4 * 32 * 16)                 //   327,680 halfs

// LDS per-wave coefficient offsets (floats)
#define CA1_O 0      // [32][16]  c0*sh0*x0
#define CA2_O 512    // [16][16]  c0*is3*(x1.sh1)
#define CX0_O 768    // [32][16]  x0 (raw, for w3/b3 path)
#define CA4_O 1280   // [16*3][16] c1*is3*sh0*x1[k]
#define CA5_O 2048   // [16*3][16] c1*is6*cross(x1,sh1)[k]
#define SH1_O 2816   // [3][16]   c1*is3*sh1[k]
#define COEF_F 2864  // floats per wave

// TDM staging: 80 groups of 2 nt-tiles; one nt-tile = 4*32*16 halfs = 4096 B
#define GROUPS      80
#define TILE_HALFS  2048
#define GRP_HALFS   (2 * TILE_HALFS)   // 4096 halfs = 8192 B per group

// A/B fragment K mapping for v_wmma_f32_16x16x32_f16:
// VGPR v (pair 2v,2v+1), lane half h:  K = 2*(v&3) + p + 8*h + 16*(v>>2)
__device__ __forceinline__ int frag_k(int v, int p, int half) {
  return 2 * (v & 3) + p + 8 * half + 16 * (v >> 2);
}

// ---------------- kernel 1: pre-swizzle W1/W2 into f16 B-fragments ----------
__global__ void prep_wfrags(const float* __restrict__ w1,
                            const float* __restrict__ w2,
                            _Float16* __restrict__ w1f,
                            _Float16* __restrict__ w2f) {
  int t = blockIdx.x * blockDim.x + threadIdx.x;
  if (t < 8 * 4 * 32) {                       // W1: 8 ntiles x 4 ktiles x 32 lanes
    int tile = t >> 5, lane = t & 31;
    int n = (tile >> 2) * 16 + (lane & 15);
    int ktb = (tile & 3) * 32;
    int half = lane >> 4;
    _Float16* dst = w1f + t * 16;
#pragma unroll
    for (int j = 0; j < 16; ++j) {
      int k = ktb + frag_k(j >> 1, j & 1, half);
      dst[j] = (_Float16)w1[k * HIDDEN + n];
    }
  } else {
    int tt = t - 8 * 4 * 32;
    if (tt >= 160 * 4 * 32) return;           // W2: 160 ntiles x 4 ktiles x 32 lanes
    int tile = tt >> 5, lane = tt & 31;
    int n = (tile >> 2) * 16 + (lane & 15);
    int ktb = (tile & 3) * 32;
    int half = lane >> 4;
    _Float16* dst = w2f + tt * 16;
#pragma unroll
    for (int j = 0; j < 16; ++j) {
      int k = ktb + frag_k(j >> 1, j & 1, half);
      dst[j] = (_Float16)w2[k * W_NUMEL + n];
    }
  }
}

// Issue a TDM copy: 8KB contiguous (one group of 2 B-fragment tiles) -> LDS.
// D# per CDNA5 ISA ch.8: group0 {count=1, lds_addr, global_addr, type=2},
// group1 {data_size=4B, tensor_dim0=2048, tensor_dim1=1, tile=2048x1, stride0=2048}.
// This toolchain exposes the 6-arg builtin:
//   (uint32x4 g0, int32x8 g1, int32x4 g2, int32x4 g3, int32x8 extra, i32 cpol)
__device__ __forceinline__ void tdm_issue(const _Float16* gsrc, unsigned lds_byte) {
  const unsigned long long ga = (unsigned long long)(uintptr_t)gsrc;
  v4u g0;
  g0.x = 1u;                                       // count=1, user mode
  g0.y = lds_byte;                                 // LDS dest (bytes)
  g0.z = (unsigned)ga;                             // global_addr[31:0]
  g0.w = (unsigned)((ga >> 32) & 0x1FFFFFFu) | (2u << 30);  // addr[56:32] | type=2
  v8i g1;
  g1[0] = 0x20000;      // data_size=2 (4 bytes); no multicast/pad/iterate
  g1[1] = 0x08000000;   // tensor_dim0[15:0]=2048 in bits[31:16]
  g1[2] = 0x00010000;   // tensor_dim0[31:16]=0 | tensor_dim1[15:0]=1
  g1[3] = 0x08000000;   // tensor_dim1[31:16]=0 | tile_dim0=2048
  g1[4] = 0x00000001;   // tile_dim1=1 | tile_dim2=0
  g1[5] = 2048;         // tensor_dim0_stride[31:0]
  g1[6] = 0;            // stride0[47:32] | stride1[15:0]
  g1[7] = 0;            // stride1[47:16]
  const v4i gz4 = {0, 0, 0, 0};
  const v8i gz8 = {0, 0, 0, 0, 0, 0, 0, 0};
  __builtin_amdgcn_tensor_load_to_lds(g0, g1, gz4, gz4, gz8, 0);
}

// ---------------- kernel 2: fused edge MLP + tensor product + scatter -------
__global__ __launch_bounds__(128, 1)
void edge_tp_kernel(const float* __restrict__ edge_attr,
                    const float* __restrict__ edge_sh,
                    const float* __restrict__ node_attr,
                    const int*   __restrict__ edge_index,
                    const float* __restrict__ fc_b1,
                    const float* __restrict__ fc_b2,
                    const _Float16* __restrict__ w1f,
                    const _Float16* __restrict__ w2f,
                    float* __restrict__ acc,
                    float* __restrict__ cnt) {
  __shared__ __align__(32) float    sCoef[4 * COEF_F];  // 45,824 B
  // time-multiplexed: phase A = f16 hidden (4 waves x 16x128), phase B = TDM
  // double buffer (2 buffers x 2 nt-tiles x 4KB).  16,384 B  (total 62,208 B)
  __shared__ __align__(32) _Float16 sDual[2 * GRP_HALFS];

  const int lane = threadIdx.x & 31;
  const int wave = threadIdx.x >> 5;
  const int tileBase = (blockIdx.x * 4 + wave) * 16;

  float*    coef = sCoef + wave * COEF_F;
  _Float16* hid  = sDual + wave * (16 * HIDDEN);
  const int cidx = lane & 15;   // column within a 16-wide tile
  const int half = lane >> 4;   // C-layout row half (e = r + 8*half)
  const int hb   = 8 * half;

  // ---- Phase 0: per-edge tensor-product coefficients into LDS -------------
  if (lane < 16) {
    const int e = lane, eg = tileBase + e;
    const int dst = edge_index[N_EDGES + eg];          // edge_dst
    const float* na = node_attr + dst * FEAT;
    const float sh0 = edge_sh[eg * 4 + 0];
    const float s1x = edge_sh[eg * 4 + 1];
    const float s1y = edge_sh[eg * 4 + 2];
    const float s1z = edge_sh[eg * 4 + 3];
#pragma unroll
    for (int u = 0; u < MUL0; ++u) {
      const float x = na[u];
      coef[CA1_O + u * 16 + e] = C0 * sh0 * x;
      coef[CX0_O + u * 16 + e] = x;
    }
#pragma unroll
    for (int u = 0; u < MUL1; ++u) {
      const float xx = na[MUL0 + u * 3 + 0];
      const float xy = na[MUL0 + u * 3 + 1];
      const float xz = na[MUL0 + u * 3 + 2];
      coef[CA2_O + u * 16 + e] = C0IS3 * (xx * s1x + xy * s1y + xz * s1z);
      coef[CA4_O + (u * 3 + 0) * 16 + e] = C1IS3 * sh0 * xx;
      coef[CA4_O + (u * 3 + 1) * 16 + e] = C1IS3 * sh0 * xy;
      coef[CA4_O + (u * 3 + 2) * 16 + e] = C1IS3 * sh0 * xz;
      coef[CA5_O + (u * 3 + 0) * 16 + e] = C1IS6 * (xy * s1z - xz * s1y);
      coef[CA5_O + (u * 3 + 1) * 16 + e] = C1IS6 * (xz * s1x - xx * s1z);
      coef[CA5_O + (u * 3 + 2) * 16 + e] = C1IS6 * (xx * s1y - xy * s1x);
    }
    coef[SH1_O + 0 * 16 + e] = C1IS3 * s1x;
    coef[SH1_O + 1 * 16 + e] = C1IS3 * s1y;
    coef[SH1_O + 2 * 16 + e] = C1IS3 * s1z;
  }

  // ---- Phase 1: load edge_attr as f16 A-fragments (K = 128 -> 4 ktiles) ---
  v16h afr[4];
  {
    const float* arow = edge_attr + (size_t)(tileBase + cidx) * EDGE_FEAT;
#pragma unroll
    for (int kt = 0; kt < 4; ++kt) {
#pragma unroll
      for (int v = 0; v < 8; ++v) {
        const int k = kt * 32 + frag_k(v, 0, half);
        const float2 f2 = *(const float2*)(arow + k);
        afr[kt][2 * v]     = (_Float16)f2.x;
        afr[kt][2 * v + 1] = (_Float16)f2.y;
      }
    }
  }
  __syncthreads();

  // ---- GEMM1: hidden = relu(E @ W1 + b1), f16 into LDS --------------------
#pragma unroll
  for (int nt = 0; nt < 8; ++nt) {
    v8f c = {};
#pragma unroll
    for (int kt = 0; kt < 4; ++kt) {
      const v16h b = *(const v16h*)(w1f + ((nt * 4 + kt) * 32 + lane) * 16);
      c = __builtin_amdgcn_wmma_f32_16x16x32_f16(false, afr[kt], false, b,
                                                 (short)0, c, false, false);
    }
    const float bv = fc_b1[nt * 16 + cidx];
#pragma unroll
    for (int r = 0; r < 8; ++r) {
      const float h = fmaxf(c[r] + bv, 0.0f);
      hid[(r + 8 * half) * HIDDEN + nt * 16 + cidx] = (_Float16)h;
    }
  }
  __syncthreads();

  // ---- Phase 2: reload hidden as A-fragments for GEMM2 --------------------
  v16h a2[4];
#pragma unroll
  for (int kt = 0; kt < 4; ++kt) {
#pragma unroll
    for (int v = 0; v < 8; ++v) {
      const int k = kt * 32 + frag_k(v, 0, half);
      const v2h hv = *(const v2h*)(hid + cidx * HIDDEN + k);
      a2[kt][2 * v]     = hv.x;
      a2[kt][2 * v + 1] = hv.y;
    }
  }
  // all waves finished reading `hid` before TDM reuses the region
  asm volatile("s_wait_dscnt 0x0" ::: "memory");
  __syncthreads();

  // ---- accumulators -------------------------------------------------------
  float o0[16];          // out0: [hi*8+r], column = cidx + 16*hi
  float o1[3][8];        // out1[k][r], column = cidx
  float b3a[8];          // b3 accumulator (w3 path)
#pragma unroll
  for (int i = 0; i < 16; ++i) o0[i] = 0.f;
#pragma unroll
  for (int k = 0; k < 3; ++k)
#pragma unroll
    for (int r = 0; r < 8; ++r) o1[k][r] = 0.f;
#pragma unroll
  for (int r = 0; r < 8; ++r) b3a[r] = 0.f;

  const unsigned ldsB = (unsigned)(uintptr_t)(sDual);  // low 32 bits = LDS offset

  // TDM prologue: prefetch groups 0,1 into the two buffers (wave 0 only;
  // tensor ops are wave-level and ignore EXEC).
  if (threadIdx.x == 0) {
    tdm_issue(w2f + 0 * GRP_HALFS, ldsB + 0 * GRP_HALFS * 2);
    tdm_issue(w2f + 1 * GRP_HALFS, ldsB + 1 * GRP_HALFS * 2);
  }

  // one 16x16 w-tile: 4 WMMAs from the LDS-staged B fragments, + b2 bias
  auto gemm_tile = [&](const _Float16* bsrc, int nt) -> v8f {
    const v16h b0 = *(const v16h*)(bsrc + (0 * 32 + lane) * 16);
    const v16h b1 = *(const v16h*)(bsrc + (1 * 32 + lane) * 16);
    const v16h b2 = *(const v16h*)(bsrc + (2 * 32 + lane) * 16);
    const v16h b3 = *(const v16h*)(bsrc + (3 * 32 + lane) * 16);
    v8f c = {};
    c = __builtin_amdgcn_wmma_f32_16x16x32_f16(false, a2[0], false, b0, (short)0, c, false, false);
    c = __builtin_amdgcn_wmma_f32_16x16x32_f16(false, a2[1], false, b1, (short)0, c, false, false);
    c = __builtin_amdgcn_wmma_f32_16x16x32_f16(false, a2[2], false, b2, (short)0, c, false, false);
    c = __builtin_amdgcn_wmma_f32_16x16x32_f16(false, a2[3], false, b3, (short)0, c, false, false);
    const float bv = fc_b2[nt * 16 + cidx];
#pragma unroll
    for (int r = 0; r < 8; ++r) c[r] += bv;
    return c;
  };

  // consume one w-tile into the tensor-product accumulators
  auto consume_tile = [&](int nt, const v8f& c) {
    if (nt < 64) {                 // w1: u = nt>>1, wcol = cidx + 16*(nt&1)
      const float* ca = coef + CA1_O + (nt >> 1) * 16 + hb;
      const float4 c0 = *(const float4*)(ca);
      const float4 c1 = *(const float4*)(ca + 4);
      float* dst = o0 + (nt & 1) * 8;
      dst[0] += c0.x * c[0]; dst[1] += c0.y * c[1];
      dst[2] += c0.z * c[2]; dst[3] += c0.w * c[3];
      dst[4] += c1.x * c[4]; dst[5] += c1.y * c[5];
      dst[6] += c1.z * c[6]; dst[7] += c1.w * c[7];
    } else if (nt < 96) {          // w2
      const float* ca = coef + CA2_O + ((nt - 64) >> 1) * 16 + hb;
      const float4 c0 = *(const float4*)(ca);
      const float4 c1 = *(const float4*)(ca + 4);
      float* dst = o0 + (nt & 1) * 8;
      dst[0] += c0.x * c[0]; dst[1] += c0.y * c[1];
      dst[2] += c0.z * c[2]; dst[3] += c0.w * c[3];
      dst[4] += c1.x * c[4]; dst[5] += c1.y * c[5];
      dst[6] += c1.z * c[6]; dst[7] += c1.w * c[7];
    } else if (nt < 128) {         // w3: b3 += x0[u] * w3
      const float* ca = coef + CX0_O + (nt - 96) * 16 + hb;
      const float4 c0 = *(const float4*)(ca);
      const float4 c1 = *(const float4*)(ca + 4);
      b3a[0] += c0.x * c[0]; b3a[1] += c0.y * c[1];
      b3a[2] += c0.z * c[2]; b3a[3] += c0.w * c[3];
      b3a[4] += c1.x * c[4]; b3a[5] += c1.y * c[5];
      b3a[6] += c1.z * c[6]; b3a[7] += c1.w * c[7];
    } else if (nt < 144) {         // w4
      const int u = nt - 128;
#pragma unroll
      for (int k = 0; k < 3; ++k) {
        const float* ca = coef + CA4_O + (u * 3 + k) * 16 + hb;
        const float4 c0 = *(const float4*)(ca);
        const float4 c1 = *(const float4*)(ca + 4);
        o1[k][0] += c0.x * c[0]; o1[k][1] += c0.y * c[1];
        o1[k][2] += c0.z * c[2]; o1[k][3] += c0.w * c[3];
        o1[k][4] += c1.x * c[4]; o1[k][5] += c1.y * c[5];
        o1[k][6] += c1.z * c[6]; o1[k][7] += c1.w * c[7];
      }
    } else {                       // w5
      const int u = nt - 144;
#pragma unroll
      for (int k = 0; k < 3; ++k) {
        const float* ca = coef + CA5_O + (u * 3 + k) * 16 + hb;
        const float4 c0 = *(const float4*)(ca);
        const float4 c1 = *(const float4*)(ca + 4);
        o1[k][0] += c0.x * c[0]; o1[k][1] += c0.y * c[1];
        o1[k][2] += c0.z * c[2]; o1[k][3] += c0.w * c[3];
        o1[k][4] += c1.x * c[4]; o1[k][5] += c1.y * c[5];
        o1[k][6] += c1.z * c[6]; o1[k][7] += c1.w * c[7];
      }
    }
  };

  // ---- GEMM2 main loop: 80 double-buffered TDM groups of 2 nt-tiles -------
  for (int g = 0; g < GROUPS; ++g) {
    if (threadIdx.x == 0) {
      if (g == GROUPS - 1) __builtin_amdgcn_s_wait_tensorcnt(0);
      else                 __builtin_amdgcn_s_wait_tensorcnt(1);
    }
    __syncthreads();                         // buffer (g&1) is ready for all waves
    const _Float16* bbase = sDual + (g & 1) * GRP_HALFS;
    {
      const v8f c0t = gemm_tile(bbase, 2 * g);
      consume_tile(2 * g, c0t);
      const v8f c1t = gemm_tile(bbase + TILE_HALFS, 2 * g + 1);
      consume_tile(2 * g + 1, c1t);
    }
    asm volatile("s_wait_dscnt 0x0" ::: "memory");
    __syncthreads();                         // everyone done reading buffer (g&1)
    if (threadIdx.x == 0 && g + 2 < GROUPS) {
      tdm_issue(w2f + (size_t)(g + 2) * GRP_HALFS, ldsB + (g & 1) * GRP_HALFS * 2);
    }
  }

  // ---- finalize w3 path: out1[e,c,k] += (c1*is3*sh1[k]) * b3[e,c] ---------
#pragma unroll
  for (int k = 0; k < 3; ++k) {
    const float* s = coef + SH1_O + k * 16 + hb;
#pragma unroll
    for (int r = 0; r < 8; ++r) o1[k][r] += s[r] * b3a[r];
  }

  // ---- scatter-add into node accumulator by edge_src ----------------------
#pragma unroll
  for (int r = 0; r < 8; ++r) {
    const int eg = tileBase + (r + 8 * half);
    const int src = edge_index[eg];                    // edge_src
    float* rowp = acc + (size_t)src * FEAT;
    atomicAdd(rowp + cidx,        o0[r]);
    atomicAdd(rowp + cidx + 16,   o0[8 + r]);
    atomicAdd(rowp + 32 + cidx * 3 + 0, o1[0][r]);
    atomicAdd(rowp + 32 + cidx * 3 + 1, o1[1][r]);
    atomicAdd(rowp + 32 + cidx * 3 + 2, o1[2][r]);
    if (cidx == 0) atomicAdd(cnt + src, 1.0f);
  }
}

// ---------------- kernel 3: mean, residual, equivariant LayerNorm ----------
__global__ __launch_bounds__(64)
void node_finalize(const float* __restrict__ acc,
                   const float* __restrict__ cnt,
                   const float* __restrict__ node_attr,
                   const float* __restrict__ ms,    // mean_shift flat [48]
                   const float* __restrict__ aw,    // affine_weight [48]
                   const float* __restrict__ ab,    // affine_bias [32]
                   float* __restrict__ out) {
  const int i = blockIdx.x * blockDim.x + threadIdx.x;
  if (i >= N_NODES) return;
  const float inv = 1.0f / fmaxf(cnt[i], 1.0f);

  float row[FEAT];
#pragma unroll
  for (int j = 0; j < FEAT; ++j)
    row[j] = acc[(size_t)i * FEAT + j] * inv + node_attr[(size_t)i * FEAT + j];

  // scalar channel (l=0)
  float mean0 = 0.f;
#pragma unroll
  for (int u = 0; u < MUL0; ++u) mean0 += row[u];
  mean0 *= (1.0f / MUL0);
  float ss0 = 0.f;
  float f0[MUL0];
#pragma unroll
  for (int u = 0; u < MUL0; ++u) {
    f0[u] = row[u] - mean0 * ms[u];
    ss0 += f0[u] * f0[u];
  }
  const float r0 = 1.0f / sqrtf(ss0 * (1.0f / MUL0) + 1e-5f);
#pragma unroll
  for (int u = 0; u < MUL0; ++u)
    out[(size_t)i * FEAT + u] = f0[u] * r0 * aw[u] + ab[u];

  // vector channel (l=1)
  float mk[3] = {0.f, 0.f, 0.f};
#pragma unroll
  for (int u = 0; u < MUL1; ++u) {
    mk[0] += row[MUL0 + u * 3 + 0];
    mk[1] += row[MUL0 + u * 3 + 1];
    mk[2] += row[MUL0 + u * 3 + 2];
  }
  mk[0] *= (1.0f / MUL1); mk[1] *= (1.0f / MUL1); mk[2] *= (1.0f / MUL1);
  float ss1 = 0.f;
#pragma unroll
  for (int u = 0; u < MUL1; ++u) {
    const float msu = ms[MUL0 + u];
#pragma unroll
    for (int k = 0; k < 3; ++k) {
      const float v = row[MUL0 + u * 3 + k] - mk[k] * msu;
      ss1 += v * v;
    }
  }
  const float r1 = 1.0f / sqrtf(ss1 * (1.0f / (MUL1 * 3)) + 1e-5f);
#pragma unroll
  for (int u = 0; u < MUL1; ++u) {
    const float msu = ms[MUL0 + u];
    const float g = r1 * aw[MUL0 + u];
#pragma unroll
    for (int k = 0; k < 3; ++k) {
      const float v = row[MUL0 + u * 3 + k] - mk[k] * msu;
      out[(size_t)i * FEAT + MUL0 + u * 3 + k] = v * g;
    }
  }
}

// ---------------- launcher --------------------------------------------------
extern "C" void kernel_launch(void* const* d_in, const int* in_sizes, int n_in,
                              void* d_out, int out_size, void* d_ws, size_t ws_size,
                              hipStream_t stream) {
  const float* node_attr = (const float*)d_in[0];
  const float* edge_attr = (const float*)d_in[1];
  const float* edge_sh   = (const float*)d_in[2];
  const float* fc_w1     = (const float*)d_in[3];
  const float* fc_b1     = (const float*)d_in[4];
  const float* fc_w2     = (const float*)d_in[5];
  const float* fc_b2     = (const float*)d_in[6];
  const float* ms        = (const float*)d_in[7];
  const float* aw        = (const float*)d_in[8];
  const float* ab        = (const float*)d_in[9];
  const int*   edge_idx  = (const int*)  d_in[10];

  float*    acc = (float*)d_ws;
  float*    cnt = (float*)((char*)d_ws + CNT_OFF);
  _Float16* w1f = (_Float16*)((char*)d_ws + W1F_OFF);
  _Float16* w2f = (_Float16*)((char*)d_ws + W2F_OFF);
  float*    out = (float*)d_out;

  // zero node accumulator + edge counts (graph-capture legal)
  (void)hipMemsetAsync(d_ws, 0, ACC_BYTES + CNT_BYTES, stream);

  // pre-swizzle weights into f16 WMMA B-fragments
  {
    const int threads = 8 * 4 * 32 + 160 * 4 * 32;   // 21504
    prep_wfrags<<<(threads + 127) / 128, 128, 0, stream>>>(fc_w1, fc_w2, w1f, w2f);
  }

  // fused edge pipeline: 4 waves/block, 16 edges/wave, 64 edges/block
  edge_tp_kernel<<<N_EDGES / 64, 128, 0, stream>>>(
      edge_attr, edge_sh, node_attr, edge_idx, fc_b1, fc_b2, w1f, w2f, acc, cnt);

  // per-node finalize
  node_finalize<<<(N_NODES + 63) / 64, 64, 0, stream>>>(
      acc, cnt, node_attr, ms, aw, ab, out);
}
